// MultiHeadAttention_39470749450859
// MI455X (gfx1250) — matrix operand order repaired and checked by reading, not compiled, not run on hardware
//
#include <hip/hip_runtime.h>
#include <math.h>

// ---------------------------------------------------------------------------
// MultiHeadAttention for MI455X (gfx1250, wave32, WMMA).
// Fixed problem: B=2, S=2048, D=1024, H=16, hd=64.
// d_out = [ out (B*S*D f32) | qk (B*H*S*S f32) ]  (reference returns a tuple).
// bf16 WMMA GEMMs (f32 accum); qk logits emitted in f32 (mandatory 512MB
// output => pipeline is HBM-bound ~80us @23.3TB/s; bf16 compute ~15us).
// This revision: double-buffered LDS (1 barrier/K-step) and CDNA5 async
// global->LDS copies (global_load_async_to_lds_b128 + s_wait_asynccnt) for
// all bf16 operand tiles; f32 tiles are software-pipelined through VGPRs
// (loads issued before the WMMA block, converted/stored after).
// Workspace layout (needs ~289 MiB):
//   [0,8M)    q  bf16 [B,H,S,hd]
//   [8M,16M)  k  bf16 [B,H,S,hd]      (== B-operand [N][K] layout for q.k^T)
//   [16M,24M) vT bf16 [B,H,hd,S]      (== B-operand [N][K] layout for A.V)
//   [24M,32M) ctx bf16 [B,S,D]
//   [32M,33M) per-row softmax stats (max, sumexp) f32x2
//   [33M,289M) masked probabilities Amod bf16 [B,H,S,S]
// ---------------------------------------------------------------------------

typedef __attribute__((ext_vector_type(16))) __bf16 v16bf;
typedef __attribute__((ext_vector_type(8)))  float  v8f;
typedef __attribute__((ext_vector_type(4)))  float  f32x4;

#define DEV static __device__ __forceinline__

constexpr int BATCH = 2;
constexpr int SEQ   = 2048;
constexpr int DIM   = 1024;
constexpr int NHEAD = 16;
constexpr int HD    = 64;
constexpr float SCALE   = 0.125f;     // 1/sqrt(64)
constexpr float NEG_INF = -1.0e9f;

constexpr int TM  = 128;   // block tile M
constexpr int TN  = 64;    // block tile N
constexpr int TK  = 32;    // K-step == WMMA K for bf16
constexpr int LDK = 40;    // padded LDS K-stride (bf16 elems) -> 80B rows

static_assert((BATCH * SEQ) % TM == 0 && DIM % TN == 0 && DIM % TK == 0, "");
static_assert(SEQ % TM == 0 && SEQ % TN == 0 && HD % TK == 0 && HD % TN == 0, "");

// CDNA5 16-bit A-matrix 16x32 per-lane layout (ISA 7.12.2):
// lanes 0-15 hold K = 0..7 and 16..23; lanes 16-31 hold K = 8..15 and 24..31.
DEV v16bf load_a_frag(const __bf16* row, int hi) {
  v16bf r;
  __builtin_memcpy(&r, row + hi * 8, 16);
  __builtin_memcpy(reinterpret_cast<char*>(&r) + 16, row + 16 + hi * 8, 16);
  return r;
}

// B-matrix: lane n (0..15) holds K=0..7 and 16..23, lane n+16 holds K=8..15 and 24..31.
DEV v16bf load_b_frag(const __bf16* row, int hi) {
  v16bf r;
  __builtin_memcpy(&r, row + hi * 8, 16);
  __builtin_memcpy(reinterpret_cast<char*>(&r) + 16, row + 16 + hi * 8, 16);
  return r;
}

DEV v8f wmma_bf16(v16bf a, v16bf b, v8f c) {
  // (neg_a, A, neg_b, B, c_mod, C, reuse_a, reuse_b)
  return __builtin_amdgcn_wmma_f32_16x16x32_bf16(false, a, false, b,
                                                 (short)0, c, false, false);
}

// CDNA5 async global->LDS copy, 16B per lane, tracked by ASYNCcnt (ISA 10.x /
// CDNA5 ISA). First operand = LDS byte address (low 32 bits of the
// generic pointer ARE the LDS offset), second = 64-bit global address, GV mode.
DEV void async_copy_b128(const void* g, void* l) {
  asm volatile("global_load_async_to_lds_b128 %0, %1, off"
               :: "v"((unsigned)(unsigned long long)l),
                  "v"((unsigned long long)g)
               : "memory");
}

DEV void wait_asynccnt0() {
  asm volatile("s_wait_asynccnt 0x0" ::: "memory");
}

enum AMode { A_F32, A_BF16 };
enum BMode { B_F32_KN, B_BF16_NK };
enum OMode { O_QKV, O_VT, O_QK, O_CTX, O_OUT };

// Generic batched WMMA GEMM: C[z] = A[z] (MxK) * B[z] (KxN as [N][K] frags).
// Double-buffered LDS; bf16 operands staged with async-to-LDS, f32 operands
// software-pipelined through VGPRs with on-the-fly bf16 conversion.
template <int AM, int BM, int OM>
__global__ __launch_bounds__(256)
void gemm_wmma_kernel(const void* __restrict__ Ap, const void* __restrict__ Bp,
                      const float* __restrict__ bias, void* __restrict__ Outp,
                      const float* __restrict__ mask,
                      int M, int N, int K,
                      long long aBatch, long long bBatch)
{
  __shared__ __bf16 ldsA[2][TM * LDK];   // [buf][row][k], padded
  __shared__ __bf16 ldsB[2][TN * LDK];   // [buf][col(n)][k], padded

  const int tid  = threadIdx.x;
  const int lane = tid & 31;
  const int wave = tid >> 5;          // 8 waves: 4 in M x 2 in N
  const int wm   = wave & 3;
  const int wn   = wave >> 2;
  const int rIW  = lane & 15;
  const int hi   = lane >> 4;

  const int m0 = blockIdx.y * TM;
  const int n0 = blockIdx.x * TN;
  const int z  = blockIdx.z;

  const float*  A32 = reinterpret_cast<const float*>(Ap);
  const __bf16* A16 = reinterpret_cast<const __bf16*>(Ap) + (long long)z * aBatch;
  const float*  B32 = reinterpret_cast<const float*>(Bp);
  const __bf16* B16 = reinterpret_cast<const __bf16*>(Bp) + (long long)z * bBatch;

  v8f acc[2][2] = {};

  // staging coordinates (256 threads)
  const int ar = tid >> 1;            // A row 0..127 (16 elems/thread)
  const int ac = (tid & 1) * 16;
  const int bk = tid & 31;            // B f32 path: k
  const int bn = (tid >> 5) * 8;      //             n base (transpose store)
  const int cn = tid >> 2;            // B bf16 path: n row 0..63
  const int cc = (tid & 3) * 8;       //              k base

  // ---- staging phases -----------------------------------------------------
  auto stage_async = [&](int kk, int bb) {   // bf16 operands: global->LDS DMA
    if constexpr (AM == A_BF16) {
      const __bf16* src = A16 + (long long)(m0 + ar) * K + kk + ac;
      async_copy_b128(src,     &ldsA[bb][ar * LDK + ac]);
      async_copy_b128(src + 8, &ldsA[bb][ar * LDK + ac + 8]);
    }
    if constexpr (BM == B_BF16_NK) {
      const __bf16* src = B16 + (long long)(n0 + cn) * K + kk + cc;
      async_copy_b128(src, &ldsB[bb][cn * LDK + cc]);
    }
  };
  auto issue_f32 = [&](int kk, f32x4* av, f32x4* bv) {  // f32 operands: loads
    if constexpr (AM == A_F32) {
      const float* src = A32 + (long long)(m0 + ar) * K + kk + ac;
      #pragma unroll
      for (int i = 0; i < 4; ++i)
        av[i] = *reinterpret_cast<const f32x4*>(src + i * 4);
      if (kk + TK < K) __builtin_prefetch(src + TK, 0, 0);
    }
    if constexpr (BM == B_F32_KN) {
      const float* src = B32 + (long long)(kk + bk) * N + n0 + bn;
      bv[0] = *reinterpret_cast<const f32x4*>(src);
      bv[1] = *reinterpret_cast<const f32x4*>(src + 4);
      if (kk + TK < K) __builtin_prefetch(src + (long long)TK * N, 0, 0);
    }
  };
  auto store_f32 = [&](int bb, const f32x4* av, const f32x4* bv) { // cvt+store
    if constexpr (AM == A_F32) {
      __bf16* dst = &ldsA[bb][ar * LDK + ac];
      #pragma unroll
      for (int i = 0; i < 4; ++i)
        #pragma unroll
        for (int j = 0; j < 4; ++j) dst[i * 4 + j] = (__bf16)av[i][j];
    }
    if constexpr (BM == B_F32_KN) {
      #pragma unroll
      for (int j = 0; j < 4; ++j) {
        ldsB[bb][(bn + j)     * LDK + bk] = (__bf16)bv[0][j];
        ldsB[bb][(bn + 4 + j) * LDK + bk] = (__bf16)bv[1][j];
      }
    }
  };
  auto sync_tile = [&]() {
    if constexpr (AM == A_BF16 || BM == B_BF16_NK) wait_asynccnt0();
    __syncthreads();
  };

  // ---- prologue: stage K-tile 0 into buffer 0 -----------------------------
  {
    f32x4 av[4], bv[2];
    issue_f32(0, av, bv);
    stage_async(0, 0);
    store_f32(0, av, bv);
    sync_tile();
  }

  // ---- main loop: compute buf, stage nxt ----------------------------------
  int buf = 0;
  for (int k0 = 0; k0 < K; k0 += TK) {
    const int  nxt  = buf ^ 1;
    const bool more = (k0 + TK) < K;

    f32x4 av[4], bv[2];
    if (more) {
      issue_f32(k0 + TK, av, bv);     // global loads in flight during WMMA
      stage_async(k0 + TK, nxt);      // async DMA runs during WMMA
    }

    v16bf a0 = load_a_frag(&ldsA[buf][(wm * 32 +      rIW) * LDK], hi);
    v16bf a1 = load_a_frag(&ldsA[buf][(wm * 32 + 16 + rIW) * LDK], hi);
    v16bf b0 = load_b_frag(&ldsB[buf][(wn * 32 +      rIW) * LDK], hi);
    v16bf b1 = load_b_frag(&ldsB[buf][(wn * 32 + 16 + rIW) * LDK], hi);

    acc[0][0] = wmma_bf16(a0, b0, acc[0][0]);
    acc[0][1] = wmma_bf16(a0, b1, acc[0][1]);
    acc[1][0] = wmma_bf16(a1, b0, acc[1][0]);
    acc[1][1] = wmma_bf16(a1, b1, acc[1][1]);

    if (more) store_f32(nxt, av, bv); // cvt+ds_store after WMMAs issued

    sync_tile();                      // asynccnt->0, then barrier
    buf = nxt;
  }

  // ---- epilogue: C layout = lane holds N=lane%16, VGPR j -> M=j+8*(lane>=16)
  #pragma unroll
  for (int t = 0; t < 2; ++t) {
    #pragma unroll
    for (int u = 0; u < 2; ++u) {
      #pragma unroll
      for (int j = 0; j < 8; ++j) {
        const int mg = m0 + wm * 32 + t * 16 + j + hi * 8;
        const int ng = n0 + wn * 32 + u * 16 + rIW;
        float v = acc[t][u][j];
        if constexpr (OM == O_QKV) {
          v += bias[ng];
          const int b = mg >> 11, s = mg & (SEQ - 1);
          const int h = ng >> 6,  d = ng & (HD - 1);
          reinterpret_cast<__bf16*>(Outp)
              [((long long)(b * NHEAD + h) * SEQ + s) * HD + d] = (__bf16)v;
        } else if constexpr (OM == O_VT) {
          v += bias[ng];
          const int b = mg >> 11, s = mg & (SEQ - 1);
          const int h = ng >> 6,  d = ng & (HD - 1);
          reinterpret_cast<__bf16*>(Outp)
              [((long long)(b * NHEAD + h) * HD + d) * SEQ + s] = (__bf16)v;
        } else if constexpr (OM == O_QK) {
          const int b = z >> 4;                     // z = b*H + h
          const float mk = mask[b * SEQ + ng];      // mask [B,1,1,S]
          v = v * SCALE + (1.0f - mk) * NEG_INF;
          reinterpret_cast<float*>(Outp)[((long long)z * SEQ + mg) * SEQ + ng] = v;
        } else if constexpr (OM == O_CTX) {
          const int b = z >> 4, h = z & (NHEAD - 1);
          reinterpret_cast<__bf16*>(Outp)
              [(long long)(b * SEQ + mg) * DIM + h * HD + ng] = (__bf16)v;
        } else { // O_OUT
          v += bias[ng];
          reinterpret_cast<float*>(Outp)[(long long)mg * DIM + ng] = v;
        }
      }
    }
  }
}

// Per-row softmax stats (max, sum of exp) over S=2048 logits. 1 block / row.
__global__ __launch_bounds__(256)
void rowstats_kernel(const float* __restrict__ qk, float2* __restrict__ stats)
{
  const long long row = blockIdx.x;
  const float* x = qk + row * SEQ;
  const int tid = threadIdx.x;
  __shared__ float red[8];

  float m = -3.0e38f;
  for (int i = tid; i < SEQ; i += 256) m = fmaxf(m, x[i]);
  #pragma unroll
  for (int o = 16; o > 0; o >>= 1) m = fmaxf(m, __shfl_xor(m, o, 32));
  if ((tid & 31) == 0) red[tid >> 5] = m;
  __syncthreads();
  float mAll = red[0];
  #pragma unroll
  for (int i = 1; i < 8; ++i) mAll = fmaxf(mAll, red[i]);
  __syncthreads();

  float s = 0.f;
  for (int i = tid; i < SEQ; i += 256) s += __expf(x[i] - mAll);
  #pragma unroll
  for (int o = 16; o > 0; o >>= 1) s += __shfl_xor(s, o, 32);
  if ((tid & 31) == 0) red[tid >> 5] = s;
  __syncthreads();
  if (tid == 0) {
    float sAll = 0.f;
    #pragma unroll
    for (int i = 0; i < 8; ++i) sAll += red[i];
    stats[row] = make_float2(mAll, sAll);
  }
}

// Amod[q][k] = softmax(qk)[q][k] * exclusive_prefix_k(softmax(qk)[q-1]);
// row 0 uses prefix == 1. One 256-thread block per row (8 contiguous cols/thr).
__global__ __launch_bounds__(256)
void colmask_kernel(const float* __restrict__ qk, const float2* __restrict__ stats,
                    void* __restrict__ AmodP)
{
  __bf16* Amod = reinterpret_cast<__bf16*>(AmodP);
  const int idx = blockIdx.x;               // (b*H + h)*S + q
  const int q   = idx & (SEQ - 1);
  const int tid = threadIdx.x;
  const int base = tid * 8;

  const float* xc = qk + (long long)idx * SEQ;
  const float2 stc = stats[idx];
  const float invZc = 1.0f / stc.y;
  __bf16* out = Amod + (long long)idx * SEQ;

  float cur[8];
  #pragma unroll
  for (int i = 0; i < 8; ++i) cur[i] = __expf(xc[base + i] - stc.x) * invZc;

  if (q == 0) {                              // uniform across block
    #pragma unroll
    for (int i = 0; i < 8; ++i) out[base + i] = (__bf16)cur[i];
    return;
  }

  const float* xp = qk + (long long)(idx - 1) * SEQ;
  const float2 stp = stats[idx - 1];
  const float invZp = 1.0f / stp.y;
  float prev[8]; float tot = 0.f;
  #pragma unroll
  for (int i = 0; i < 8; ++i) {
    prev[i] = __expf(xp[base + i] - stp.x) * invZp;
    tot += prev[i];
  }

  // exclusive scan of per-thread totals (Hillis-Steele over 256 entries)
  __shared__ float sc[256];
  sc[tid] = tot;
  __syncthreads();
  for (int off = 1; off < 256; off <<= 1) {
    float v = (tid >= off) ? sc[tid - off] : 0.f;
    __syncthreads();
    sc[tid] += v;
    __syncthreads();
  }
  float run = sc[tid] - tot;                 // exclusive prefix at col base
  #pragma unroll
  for (int i = 0; i < 8; ++i) {
    out[base + i] = (__bf16)(cur[i] * run);
    run += prev[i];
  }
}

extern "C" void kernel_launch(void* const* d_in, const int* in_sizes, int n_in,
                              void* d_out, int out_size, void* d_ws, size_t ws_size,
                              hipStream_t stream)
{
  (void)in_sizes; (void)n_in; (void)out_size; (void)ws_size;

  const float* query  = (const float*)d_in[0];
  const float* keys   = (const float*)d_in[1];
  const float* values = (const float*)d_in[2];
  const float* mask   = (const float*)d_in[3];
  const float* wq_k   = (const float*)d_in[4];
  const float* wq_b   = (const float*)d_in[5];
  const float* wk_k   = (const float*)d_in[6];
  const float* wk_b   = (const float*)d_in[7];
  const float* wv_k   = (const float*)d_in[8];
  const float* wv_b   = (const float*)d_in[9];
  const float* wo_k   = (const float*)d_in[10];
  const float* wo_b   = (const float*)d_in[11];

  float* out = (float*)d_out;                               // [B,S,D]
  float* qk  = out + (long long)BATCH * SEQ * DIM;          // [B,H,S,S]

  char* ws = (char*)d_ws;
  void*   q_bf   = (void*)(ws);
  void*   k_bf   = (void*)(ws + (8ll  << 20));
  void*   vT_bf  = (void*)(ws + (16ll << 20));
  void*   ctx_bf = (void*)(ws + (24ll << 20));
  float2* stats  = (float2*)(ws + (32ll << 20));
  void*   Amod   = (void*)(ws + (33ll << 20));

  const dim3 blk(256);
  const dim3 gProj(DIM / TN, (BATCH * SEQ) / TM, 1);        // (16, 32, 1)
  const dim3 gQK (SEQ / TN, SEQ / TM, BATCH * NHEAD);       // (32, 16, 32)
  const dim3 gAV (HD  / TN, SEQ / TM, BATCH * NHEAD);       // ( 1, 16, 32)

  // projections (f32 in, bf16 out; K scatter == B-layout for q.k^T, V transposed)
  gemm_wmma_kernel<A_F32, B_F32_KN, O_QKV><<<gProj, blk, 0, stream>>>(
      query,  wq_k, wq_b, q_bf,  nullptr, BATCH*SEQ, DIM, DIM, 0, 0);
  gemm_wmma_kernel<A_F32, B_F32_KN, O_QKV><<<gProj, blk, 0, stream>>>(
      keys,   wk_k, wk_b, k_bf,  nullptr, BATCH*SEQ, DIM, DIM, 0, 0);
  gemm_wmma_kernel<A_F32, B_F32_KN, O_VT ><<<gProj, blk, 0, stream>>>(
      values, wv_k, wv_b, vT_bf, nullptr, BATCH*SEQ, DIM, DIM, 0, 0);

  // qk logits (scaled + masked) -> d_out second region, f32
  gemm_wmma_kernel<A_BF16, B_BF16_NK, O_QK><<<gQK, blk, 0, stream>>>(
      q_bf, k_bf, nullptr, qk, mask, SEQ, SEQ, HD,
      (long long)SEQ * HD, (long long)SEQ * HD);

  // softmax stats + colinear-masked probabilities (bf16)
  rowstats_kernel<<<BATCH * NHEAD * SEQ, blk, 0, stream>>>(qk, stats);
  colmask_kernel <<<BATCH * NHEAD * SEQ, blk, 0, stream>>>(qk, stats, Amod);

  // context = Amod @ V  (V already [hd][S] => B-operand layout)
  gemm_wmma_kernel<A_BF16, B_BF16_NK, O_CTX><<<gAV, blk, 0, stream>>>(
      Amod, vT_bf, nullptr, ctx_bf, nullptr, SEQ, HD, SEQ,
      (long long)SEQ * SEQ, (long long)HD * SEQ);

  // output projection -> d_out first region, f32
  gemm_wmma_kernel<A_BF16, B_F32_KN, O_OUT><<<gProj, blk, 0, stream>>>(
      ctx_bf, wo_k, wo_b, out, nullptr, BATCH*SEQ, DIM, DIM, 0, 0);
}
